// LSTMModel1_73821897884293
// MI455X (gfx1250) — compile-verified, matrix operand order checked by reading, not run on hardware
//
#include <hip/hip_runtime.h>

#define N_B    64
#define L_T    1024
#define IN_D   64
#define H1_D   256
#define H2_D   128
#define G4     1024          // 4*H1
#define KSTEPS 10            // (IN_D + H1_D)/32
#define KSTEPS_H 8           // H1_D/32
#define NT_H   8             // H2_D/16
#define AXS    328           // padded row stride (ushorts) for [x_t | h] LDS rows
#define GSTRIDE 1028         // padded row stride (floats) for gate scratch

typedef __attribute__((ext_vector_type(16))) __bf16 v16bf;
typedef __attribute__((ext_vector_type(8)))  float  v8f;

union Frag16 { v16bf v; uint4 q[2]; };

__device__ __forceinline__ unsigned short f2bf(float f) {
  unsigned u = __float_as_uint(f);
  u += 0x7FFFu + ((u >> 16) & 1u);            // round-to-nearest-even
  return (unsigned short)(u >> 16);
}

__device__ __forceinline__ float sigm(float x) {
  return 1.0f / (1.0f + __expf(-x));
}

// ---------------------------------------------------------------------------
// Prep: pack combined [W_ih | W_hh] (K=320 x N=1024) and W1 (K=256 x N=128)
// into bf16 WMMA-B fragment order: [ntile][kstep][lane][16 halves], so each
// lane fetches its whole fragment with one contiguous 32B load. Also b_ih+b_hh.
// ---------------------------------------------------------------------------
__global__ void prep_pack(const float* __restrict__ Wih, const float* __restrict__ Whh,
                          const float* __restrict__ bih, const float* __restrict__ bhh,
                          const float* __restrict__ W1,
                          unsigned short* __restrict__ bpack,
                          unsigned short* __restrict__ w1pack,
                          float* __restrict__ gbias) {
  const int BP = (G4 / 16) * KSTEPS * 32 * 16;    // 327680
  const int WP = NT_H * KSTEPS_H * 32 * 16;       // 32768
  int tid = blockIdx.x * blockDim.x + threadIdx.x;
  if (tid < BP) {
    int j = tid & 15, lane = (tid >> 4) & 31, rem = tid >> 9;
    int k = rem % KSTEPS, nt = rem / KSTEPS;
    int koff = (j < 8 ? j : j + 8) + ((lane >> 4) << 3);
    int kk = k * 32 + koff;
    int nc = nt * 16 + (lane & 15);
    float v = (kk < IN_D) ? Wih[nc * IN_D + kk] : Whh[nc * H1_D + (kk - IN_D)];
    bpack[tid] = f2bf(v);
  } else if (tid < BP + WP) {
    int t2 = tid - BP;
    int j = t2 & 15, lane = (t2 >> 4) & 31, rem = t2 >> 9;
    int k = rem % KSTEPS_H, nt = rem / KSTEPS_H;
    int koff = (j < 8 ? j : j + 8) + ((lane >> 4) << 3);
    int kk = k * 32 + koff;
    int nc = nt * 16 + (lane & 15);
    w1pack[t2] = f2bf(W1[nc * H1_D + kk]);
  } else if (tid < BP + WP + G4) {
    int g = tid - BP - WP;
    gbias[g] = bih[g] + bhh[g];
  }
}

// ---------------------------------------------------------------------------
// Persistent fused LSTM + MLP head. One workgroup = 16 batch rows (one WMMA
// M-tile), 16 waves. Per step: 16 waves x 4 N-tiles x 10 k-steps of
// v_wmma_f32_16x16x32_bf16 for the gates. The packed-B weight stream stays in
// L2; an opaque per-iteration SGPR offset (not an opaque pointer, which would
// degrade the loads to FLAT) blocks LICM from hoisting 320 VGPRs of fragments
// into scratch spills while keeping GLOBAL_LOAD lowering. fp32 activations;
// waves 0-7 then run the head GEMM on the fresh h (head B fragments DO stay
// hoisted in registers) while waves 8-15 prefetch x_{t+1}.
// ---------------------------------------------------------------------------
__global__ __launch_bounds__(512)
void lstm_fused(const float* __restrict__ x,
                const unsigned short* __restrict__ bpack,
                const unsigned short* __restrict__ w1pack,
                const float* __restrict__ gbias,
                const float* __restrict__ b1g,
                const float* __restrict__ W2g,
                const float* __restrict__ b2g,
                float* __restrict__ out,
                float* __restrict__ cs) {
  __shared__ unsigned short ax[16 * AXS];   // [x_t (64) | h (256)] per row, bf16
  __shared__ float gsm[16 * GSTRIDE];       // gate pre-activations, fp32
  __shared__ float hpart[NT_H][16];         // head per-tile row partials

  const int tid  = threadIdx.x;
  const int lane = tid & 31;
  const int wv   = tid >> 5;      // wave 0..15
  const int lm   = lane & 15;
  const int lhi  = lane >> 4;
  const int r0   = blockIdx.x * 16;

  for (int i = tid; i < 16 * AXS; i += 512) ax[i] = 0;   // h0 = 0

  const int em = tid >> 5;          // elementwise row
  const int ej = (tid & 31) * 8;    // elementwise j base
  float cprev[8];
#pragma unroll
  for (int e = 0; e < 8; ++e) cprev[e] = 0.f;

  float biasv[4];
#pragma unroll
  for (int tt = 0; tt < 4; ++tt) biasv[tt] = gbias[(wv * 4 + tt) * 16 + lm];

  float b1v = 0.f, w2v = 0.f;
  if (wv < NT_H) { int nc = wv * 16 + lm; b1v = b1g[nc]; w2v = W2g[nc]; }
  const float b2v = b2g[0];

  { // load x_0
    int m = tid >> 5, i2 = (tid & 31) * 2;
    const float2 v = *(const float2*)(x + (size_t)(r0 + m) * (L_T * IN_D) + i2);
    ax[m * AXS + i2]     = f2bf(v.x);
    ax[m * AXS + i2 + 1] = f2bf(v.y);
  }

  // Opaque zero offset: defeats LICM hoist->spill of the B stream while
  // preserving the global address space of bpack (GLOBAL_LOAD, not FLAT).
  int zoff = 0;

  for (int t = 0; t < L_T; ++t) {
    __syncthreads();   // ax holds [x_t | h_t]

    asm volatile("" : "+s"(zoff));           // re-opaque every iteration
    const uint4* bbase = (const uint4*)bpack + zoff;

    // ---- gates = [x_t | h] @ Wcomb^T + (b_ih + b_hh) ----
    v8f acc[4];
#pragma unroll
    for (int tt = 0; tt < 4; ++tt)
#pragma unroll
      for (int r = 0; r < 8; ++r) acc[tt][r] = biasv[tt];

#pragma unroll
    for (int k = 0; k < KSTEPS; ++k) {
      Frag16 af;
      const unsigned short* rp = &ax[lm * AXS + k * 32 + lhi * 8];
      af.q[0] = *(const uint4*)rp;
      af.q[1] = *(const uint4*)(rp + 16);
#pragma unroll
      for (int tt = 0; tt < 4; ++tt) {
        const int nt = wv * 4 + tt;
        const uint4* bp = bbase + ((size_t)(nt * KSTEPS + k) * 32 + lane) * 2;
        Frag16 bf;
        bf.q[0] = bp[0];
        bf.q[1] = bp[1];
        acc[tt] = __builtin_amdgcn_wmma_f32_16x16x32_bf16(false, af.v, false, bf.v,
                                                          (short)0, acc[tt], false, false);
      }
    }

#pragma unroll
    for (int tt = 0; tt < 4; ++tt) {
      const int nc = (wv * 4 + tt) * 16 + lm;
#pragma unroll
      for (int r = 0; r < 8; ++r)
        gsm[(r + lhi * 8) * GSTRIDE + nc] = acc[tt][r];
    }
    __syncthreads();

    // ---- elementwise LSTM cell (fp32), write c to global, h(bf16) to LDS ----
    {
      const float* gi = &gsm[em * GSTRIDE + ej];
      float hreg[8];
#pragma unroll
      for (int e = 0; e < 8; ++e) {
        float iv = sigm(gi[e]);
        float fv = sigm(gi[e + 256]);
        float gv = tanhf(gi[e + 512]);
        float ov = sigm(gi[e + 768]);
        float cv = fv * cprev[e] + iv * gv;
        cprev[e] = cv;
        hreg[e] = ov * tanhf(cv);
      }
      float4* cp = (float4*)(cs + (size_t)(r0 + em) * (L_T * H1_D) + (size_t)t * H1_D + ej);
      cp[0] = make_float4(cprev[0], cprev[1], cprev[2], cprev[3]);
      cp[1] = make_float4(cprev[4], cprev[5], cprev[6], cprev[7]);
      unsigned short* hp = &ax[em * AXS + IN_D + ej];
#pragma unroll
      for (int e = 0; e < 8; ++e) hp[e] = f2bf(hreg[e]);
    }
    __syncthreads();

    // ---- head GEMM on fresh h (waves 0-7); x_{t+1} prefetch (waves 8-15) ----
    if (wv < NT_H) {
      v8f hacc;
#pragma unroll
      for (int r = 0; r < 8; ++r) hacc[r] = b1v;
#pragma unroll
      for (int k = 0; k < KSTEPS_H; ++k) {
        Frag16 af;
        const unsigned short* rp = &ax[lm * AXS + IN_D + k * 32 + lhi * 8];
        af.q[0] = *(const uint4*)rp;
        af.q[1] = *(const uint4*)(rp + 16);
        const uint4* bp = (const uint4*)w1pack + ((size_t)(wv * KSTEPS_H + k) * 32 + lane) * 2;
        Frag16 bf;
        bf.q[0] = bp[0];
        bf.q[1] = bp[1];
        hacc = __builtin_amdgcn_wmma_f32_16x16x32_bf16(false, af.v, false, bf.v,
                                                       (short)0, hacc, false, false);
      }
      // relu(.)*W2, reduce across the 16 N-columns of this tile
#pragma unroll
      for (int r = 0; r < 8; ++r) {
        float v = hacc[r];
        v = v > 0.f ? v * w2v : 0.f;
        v += __shfl_xor(v, 1);
        v += __shfl_xor(v, 2);
        v += __shfl_xor(v, 4);
        v += __shfl_xor(v, 8);
        if (lm == 0) hpart[wv][r + lhi * 8] = v;
      }
    } else if (t + 1 < L_T) {
      int lt = tid - 256;
      int m = lt >> 4, i4 = (lt & 15) * 4;
      const float* xr = x + (size_t)(r0 + m) * (L_T * IN_D) + (size_t)(t + 1) * IN_D + i4;
      const float4 v = *(const float4*)xr;
      if (t + 2 < L_T) __builtin_prefetch(xr + IN_D, 0, 0);  // global_prefetch_b8
      unsigned short* axp = &ax[m * AXS + i4];
      axp[0] = f2bf(v.x); axp[1] = f2bf(v.y); axp[2] = f2bf(v.z); axp[3] = f2bf(v.w);
    }
    __syncthreads();

    if (tid < 16) {
      float s = b2v;
#pragma unroll
      for (int q = 0; q < NT_H; ++q) s += hpart[q][tid];
      out[(size_t)(r0 + tid) * L_T + t] = s;
    }
  }
}

// ---------------------------------------------------------------------------
extern "C" void kernel_launch(void* const* d_in, const int* in_sizes, int n_in,
                              void* d_out, int out_size, void* d_ws, size_t ws_size,
                              hipStream_t stream) {
  (void)in_sizes; (void)n_in; (void)out_size; (void)ws_size;
  const float* x   = (const float*)d_in[0];
  const float* Wih = (const float*)d_in[1];
  const float* Whh = (const float*)d_in[2];
  const float* bih = (const float*)d_in[3];
  const float* bhh = (const float*)d_in[4];
  const float* W1  = (const float*)d_in[5];
  const float* b1  = (const float*)d_in[6];
  const float* W2  = (const float*)d_in[7];
  const float* b2  = (const float*)d_in[8];

  unsigned short* bpack  = (unsigned short*)d_ws;                       // 655360 B
  unsigned short* w1pack = (unsigned short*)((char*)d_ws + 655360);     // 65536 B
  float*          gbias  = (float*)((char*)d_ws + 720896);              // 4096 B

  float* out = (float*)d_out;                      // [N, L, 1]
  float* cs  = (float*)d_out + (size_t)N_B * L_T;  // [N, L, H1]

  const int total = 327680 + 32768 + 1024;
  prep_pack<<<(total + 255) / 256, 256, 0, stream>>>(Wih, Whh, bih, bhh, W1,
                                                     bpack, w1pack, gbias);
  lstm_fused<<<N_B / 16, 512, 0, stream>>>(x, bpack, w1pack, gbias, b1, W2, b2, out, cs);
}
